// MHAttention_GAT_FC_90280212562514
// MI455X (gfx1250) — compile-verified, matrix operand order checked by reading, not run on hardware
//
#include <hip/hip_runtime.h>
#include <hip/hip_bf16.h>

// ---------------------------------------------------------------------------
// MI455X (gfx1250): transformer encoder (2x MHA+LN) -> masked mean pool ->
// 2x GATConv -> FC.  f16 end-to-end in global memory, f32 accumulate via
// v_wmma_f32_16x16x32_f16.  GEMM: double-buffered LDS tiles, A staged with
// global_load_async_to_lds_b128 (ASYNCcnt) overlapping the WMMA compute.
// Fragments are preloaded in batches ahead of each WMMA chain so DS latency
// overlaps WMMA issue instead of serializing on s_wait_dscnt 0.
// ---------------------------------------------------------------------------

typedef __attribute__((ext_vector_type(16))) _Float16 v16h;
typedef __attribute__((ext_vector_type(8)))  _Float16 v8h;
typedef __attribute__((ext_vector_type(8)))  float    v8f;
typedef __attribute__((ext_vector_type(4)))  int      v4i;

#define NB    1024
#define NL    128
#define NIN   64
#define NE    256
#define NQKV  768
#define NHEAD 4
#define NEDGE 16384

__device__ __forceinline__ v8f wmma_f16(v16h a, v16h b, v8f c) {
  return __builtin_amdgcn_wmma_f32_16x16x32_f16(false, a, false, b, (short)0, c,
                                                false, false);
}

// 16-half fragment from two contiguous 16B chunks (ISA 7.12.2 K striping).
__device__ __forceinline__ v16h frag16(const _Float16* row, int hi) {
  v8h lo = *(const v8h*)(row + hi * 8);
  v8h hv = *(const v8h*)(row + 16 + hi * 8);
  v16h f;
#pragma unroll
  for (int i = 0; i < 8; ++i) { f[i] = lo[i]; f[8 + i] = hv[i]; }
  return f;
}

// 16B global -> LDS copy.  CDNA5 async DMA (tracked by ASYNCcnt); the
// builtin's parameters are int4-vector pointers in AS1 (global) / AS3 (LDS).
typedef __attribute__((address_space(1))) v4i* as1_v4i;
typedef __attribute__((address_space(3))) v4i* as3_v4i;

__device__ __forceinline__ void cp16(const _Float16* g, _Float16* l) {
#if __has_builtin(__builtin_amdgcn_global_load_async_to_lds_b128)
  __builtin_amdgcn_global_load_async_to_lds_b128(
      (as1_v4i)(const void*)(g), (as3_v4i)(void*)(l), 0, 0);
#else
  *(v8h*)l = *(const v8h*)g;
#endif
}
__device__ __forceinline__ void cp16_wait() {
#if __has_builtin(__builtin_amdgcn_global_load_async_to_lds_b128)
#if __has_builtin(__builtin_amdgcn_s_wait_asynccnt)
  __builtin_amdgcn_s_wait_asynccnt(0);
#else
  asm volatile("s_wait_asynccnt 0" ::: "memory");
#endif
#endif
}

// ---------------------------------------------------------------------------
// f32 -> f16 conversion (weights once per launch, and the network input x)
// ---------------------------------------------------------------------------
__global__ __launch_bounds__(256) void k_cvt(const float* __restrict__ s,
                                             _Float16* __restrict__ d, int n) {
  int i = blockIdx.x * 256 + threadIdx.x;
  if (i < n) d[i] = (_Float16)s[i];
}

// ---------------------------------------------------------------------------
// GEMM: C[M,N] = A[M,K] * W[K,N] (+ bias).  A,W f16; acc f32.
// Block tile 256x64, wave tile 32x64, K-step 32, double-buffered LDS.
// NARROW: N < 64 (clamped scalar B staging, guarded stores).
// OUTF16: C stored as f16 (else f32).  M % 256 == 0, K % 32 == 0.
// ---------------------------------------------------------------------------
#define GPAD 48

template <bool NARROW, bool OUTF16>
__global__ __launch_bounds__(256) void k_gemm16(
    const _Float16* __restrict__ A, const _Float16* __restrict__ W,
    const float* __restrict__ bias, void* __restrict__ Cp,
    int M, int N, int K) {
  __shared__ _Float16 Al[2][256][GPAD];  // 2 x 24 KB, [row][k]
  __shared__ _Float16 Bt[2][64][GPAD];   // 2 x  6 KB, [n][k] (B transposed)

  const int t    = threadIdx.x;
  const int lane = t & 31;
  const int wv   = t >> 5;
  const int m    = lane & 15;
  const int hi   = lane >> 4;
  const int gm0  = blockIdx.x * 256;
  const int n0   = blockIdx.y * 64;

  auto stageA = [&](int k0, int buf) {
#pragma unroll
    for (int j = 0; j < 4; ++j) {
      int idx = j * 256 + t;       // 1024 16B chunks over 256x32 halves
      int row = idx >> 2;
      int c8  = idx & 3;
      cp16(A + (size_t)(gm0 + row) * K + k0 + c8 * 8, &Al[buf][row][c8 * 8]);
    }
  };
  auto stageB = [&](int k0, int buf) {
    if (!NARROW) {
      int row = t >> 3;            // 8 chunks per 64-half row
      int c8  = t & 7;
      v8h wv8 = *(const v8h*)(W + (size_t)(k0 + row) * N + n0 + c8 * 8);
#pragma unroll
      for (int e = 0; e < 8; ++e) Bt[buf][c8 * 8 + e][row] = wv8[e];
    } else {
#pragma unroll
      for (int j = 0; j < 8; ++j) {
        int idx = j * 256 + t;
        int row = idx >> 6;
        int c   = idx & 63;
        int n   = n0 + c;
        Bt[buf][c][row] =
            (n < N) ? W[(size_t)(k0 + row) * N + n] : (_Float16)0.0f;
      }
    }
  };

  const int nsteps = K >> 5;
  v8f acc[2][4] = {};

  stageA(0, 0);
  stageB(0, 0);
  cp16_wait();
  __syncthreads();

  for (int i = 0; i < nsteps; ++i) {
    const int cur = i & 1;
    if (i + 1 < nsteps) {            // prefetch next tile into other buffer
      stageA((i + 1) << 5, 1 - cur); // async DMA overlaps compute below
      stageB((i + 1) << 5, 1 - cur);
    }
    // preload all 6 fragments, then issue the 8-WMMA chain uninterrupted
    v16h bfr[4], afr[2];
#pragma unroll
    for (int nt = 0; nt < 4; ++nt)
      bfr[nt] = frag16(&Bt[cur][nt * 16 + m][0], hi);
#pragma unroll
    for (int ms = 0; ms < 2; ++ms)
      afr[ms] = frag16(&Al[cur][wv * 32 + ms * 16 + m][0], hi);
#pragma unroll
    for (int ms = 0; ms < 2; ++ms)
#pragma unroll
      for (int nt = 0; nt < 4; ++nt)
        acc[ms][nt] = wmma_f16(afr[ms], bfr[nt], acc[ms][nt]);
    if (i + 1 < nsteps) {
      cp16_wait();
      __syncthreads();
    }
  }

#pragma unroll
  for (int ms = 0; ms < 2; ++ms) {
#pragma unroll
    for (int nt = 0; nt < 4; ++nt) {
      int col = n0 + nt * 16 + m;
      if (!NARROW || col < N) {
        float bv = bias ? bias[col] : 0.0f;
#pragma unroll
        for (int r = 0; r < 8; ++r) {
          int row = gm0 + wv * 32 + ms * 16 + (hi ? r + 8 : r);
          float o = acc[ms][nt][r] + bv;
          if (OUTF16) ((_Float16*)Cp)[(size_t)row * N + col] = (_Float16)o;
          else        ((float*)Cp)[(size_t)row * N + col] = o;
        }
      }
    }
  }
}

// ---------------------------------------------------------------------------
// Fused attention: one block per (batch, head).  qkv16: [B, L, 768] f16.
// K async-staged row-major (stride 72), V staged transposed (stride 136);
// probs overlap the K region.  Wave -> 16 query rows.
// ---------------------------------------------------------------------------
#define KSTR 72
#define PSTR 136

__global__ __launch_bounds__(256) void k_attention(
    const _Float16* __restrict__ qkv, const int* __restrict__ lengths,
    _Float16* __restrict__ out) {
  __shared__ _Float16 smem0[8 * 16 * PSTR];  // Kl (128x72) then P (128x136)
  __shared__ _Float16 Vt[64][PSTR];
  _Float16* Kl = smem0;
  _Float16* P  = smem0;

  const int t    = threadIdx.x;
  const int lane = t & 31;
  const int wv   = t >> 5;
  const int m    = lane & 15;
  const int hi   = lane >> 4;
  const int b    = blockIdx.x >> 2;
  const int hd   = blockIdx.x & 3;
  const int m0   = wv * 16;
  const int LD   = NQKV;

  const _Float16* base = qkv + (size_t)b * NL * NQKV;
  const _Float16* Q  = base + hd * 64;
  const _Float16* Km = base + NE + hd * 64;
  const _Float16* Vm = base + 2 * NE + hd * 64;

  // K: async DMA into padded rows (overlaps the manual V transpose below).
#pragma unroll
  for (int j = 0; j < 4; ++j) {
    int idx = j * 256 + t;       // 1024 chunks over 128x64 halves
    int row = idx >> 3;
    int c8  = idx & 7;
    cp16(Km + (size_t)row * LD + c8 * 8, &Kl[row * KSTR + c8 * 8]);
  }
  v8h vbuf[4];
#pragma unroll
  for (int j = 0; j < 4; ++j) {
    int idx = j * 256 + t;
    int row = idx >> 3;
    int c8  = idx & 7;
    vbuf[j] = *(const v8h*)(Vm + (size_t)row * LD + c8 * 8);
  }
#pragma unroll
  for (int j = 0; j < 4; ++j) {
    int idx = j * 256 + t;
    int row = idx >> 3;
    int c8  = idx & 7;
#pragma unroll
    for (int e = 0; e < 8; ++e) Vt[c8 * 8 + e][row] = vbuf[j][e];
  }
  cp16_wait();
  __syncthreads();

  // ---- S = Q K^T (preload all 8 K fragments, then 8-WMMA chain) ---------
  v8f s[8] = {};
#pragma unroll
  for (int k0 = 0; k0 < 64; k0 += 32) {
    v16h a = frag16(Q + (size_t)(m0 + m) * LD + k0, hi);
    v16h kfr[8];
#pragma unroll
    for (int nt = 0; nt < 8; ++nt)
      kfr[nt] = frag16(&Kl[(nt * 16 + m) * KSTR + k0], hi);
#pragma unroll
    for (int nt = 0; nt < 8; ++nt)
      s[nt] = wmma_f16(a, kfr[nt], s[nt]);
  }

  // ---- scale + pad mask + softmax ---------------------------------------
  const int len = lengths[b];
#pragma unroll
  for (int r = 0; r < 8; ++r) {
    float vmax = -3.0e38f;
#pragma unroll
    for (int nt = 0; nt < 8; ++nt) {
      int col = nt * 16 + m;
      float v = s[nt][r] * 0.125f;
      if (col >= len) v = -1.0e9f;
      s[nt][r] = v;
      vmax = fmaxf(vmax, v);
    }
#pragma unroll
    for (int off = 1; off <= 8; off <<= 1)
      vmax = fmaxf(vmax, __shfl_xor(vmax, off, 32));
    float sum = 0.0f;
#pragma unroll
    for (int nt = 0; nt < 8; ++nt) {
      float p = __expf(s[nt][r] - vmax);
      s[nt][r] = p;
      sum += p;
    }
#pragma unroll
    for (int off = 1; off <= 8; off <<= 1)
      sum += __shfl_xor(sum, off, 32);
    float inv = 1.0f / sum;
#pragma unroll
    for (int nt = 0; nt < 8; ++nt) s[nt][r] *= inv;
  }
  __syncthreads();   // all waves done with Kl before P overwrites it

#pragma unroll
  for (int r = 0; r < 8; ++r) {
    int row = hi ? r + 8 : r;
#pragma unroll
    for (int nt = 0; nt < 8; ++nt)
      P[(m0 + row) * PSTR + nt * 16 + m] = (_Float16)s[nt][r];
  }
  __syncthreads();

  // ---- O = P V (preload 4 V fragments per K-step) -----------------------
  v8f o[4] = {};
#pragma unroll
  for (int kk = 0; kk < 128; kk += 32) {
    v16h a = frag16(&P[(m0 + m) * PSTR + kk], hi);
    v16h vfr[4];
#pragma unroll
    for (int nt = 0; nt < 4; ++nt)
      vfr[nt] = frag16(&Vt[nt * 16 + m][kk], hi);
#pragma unroll
    for (int nt = 0; nt < 4; ++nt)
      o[nt] = wmma_f16(a, vfr[nt], o[nt]);
  }
#pragma unroll
  for (int nt = 0; nt < 4; ++nt) {
#pragma unroll
    for (int r = 0; r < 8; ++r) {
      int row = m0 + (hi ? r + 8 : r);
      out[((size_t)b * NL + row) * NE + hd * 64 + nt * 16 + m] =
          (_Float16)o[nt][r];
    }
  }
}

// ---------------------------------------------------------------------------
// y16 = [relu]( LN( x (+ res) (+ colbias) ) * g + b ),  D = 256.
// x is f16 or f32 (one pointer null).  Wave per row; lane owns 8 columns.
// ---------------------------------------------------------------------------
__global__ __launch_bounds__(256) void k_res_ln(
    const _Float16* __restrict__ x16, const float* __restrict__ x32,
    const _Float16* __restrict__ res16, const float* __restrict__ colbias,
    const float* __restrict__ g, const float* __restrict__ bta,
    _Float16* __restrict__ y, int rows, int relu) {
  const int lane = threadIdx.x & 31;
  const int wv   = threadIdx.x >> 5;
  const int row  = blockIdx.x * 8 + wv;
  if (row >= rows) return;

  const int c0 = lane * 8;
  float v[8];
  float sum = 0.0f, sq = 0.0f;
  if (x16) {
    v8h xv = *(const v8h*)(x16 + (size_t)row * NE + c0);
#pragma unroll
    for (int j = 0; j < 8; ++j) v[j] = (float)xv[j];
  } else {
#pragma unroll
    for (int q = 0; q < 2; ++q) {
      float4 xv = *(const float4*)(x32 + (size_t)row * NE + c0 + q * 4);
      v[q * 4 + 0] = xv.x; v[q * 4 + 1] = xv.y;
      v[q * 4 + 2] = xv.z; v[q * 4 + 3] = xv.w;
    }
  }
  if (res16) {
    v8h rv = *(const v8h*)(res16 + (size_t)row * NE + c0);
#pragma unroll
    for (int j = 0; j < 8; ++j) v[j] += (float)rv[j];
  }
  if (colbias) {
#pragma unroll
    for (int q = 0; q < 2; ++q) {
      float4 bv = *(const float4*)(colbias + c0 + q * 4);
      v[q * 4 + 0] += bv.x; v[q * 4 + 1] += bv.y;
      v[q * 4 + 2] += bv.z; v[q * 4 + 3] += bv.w;
    }
  }
#pragma unroll
  for (int j = 0; j < 8; ++j) { sum += v[j]; sq += v[j] * v[j]; }
#pragma unroll
  for (int off = 1; off <= 16; off <<= 1) {
    sum += __shfl_xor(sum, off, 32);
    sq  += __shfl_xor(sq,  off, 32);
  }
  float mean = sum * (1.0f / NE);
  float var  = sq * (1.0f / NE) - mean * mean;
  float rstd = rsqrtf(var + 1e-5f);
  v8h o8;
#pragma unroll
  for (int j = 0; j < 8; ++j) {
    int c = c0 + j;
    float o = (v[j] - mean) * rstd * g[c] + bta[c];
    if (relu) o = fmaxf(o, 0.0f);
    o8[j] = (_Float16)o;
  }
  *(v8h*)(y + (size_t)row * NE + c0) = o8;
}

// ---------------------------------------------------------------------------
// Masked mean pooling (f16 in/out, f32 accumulate)
// ---------------------------------------------------------------------------
__global__ __launch_bounds__(256) void k_pool(
    const _Float16* __restrict__ h, const int* __restrict__ lengths,
    _Float16* __restrict__ pooled) {
  const int b = blockIdx.x;
  const int f = threadIdx.x;
  const int len = lengths[b];
  float sum = 0.0f;
  for (int l = 0; l < len; ++l)
    sum += (float)h[((size_t)b * NL + l) * NE + f];
  pooled[(size_t)b * NE + f] = (_Float16)(sum / (float)len);
}

// ---------------------------------------------------------------------------
// GAT (scatter-bound; f32 scalar path)
// ---------------------------------------------------------------------------
__device__ __forceinline__ void atomicMaxF(float* addr, float val) {
  unsigned* ua = (unsigned*)addr;
  unsigned old = *ua;
  while (__uint_as_float(old) < val) {
    unsigned assumed = old;
    old = atomicCAS(ua, assumed, __float_as_uint(val));
    if (old == assumed) break;
  }
}

__device__ __forceinline__ float leaky02(float e) {
  return e > 0.0f ? e : 0.2f * e;
}

__global__ __launch_bounds__(256) void k_gat_scores(
    const float* __restrict__ xh, const float* __restrict__ asrc,
    const float* __restrict__ adst, float* __restrict__ sb,
    float* __restrict__ db) {
  int idx = blockIdx.x * 256 + threadIdx.x;
  if (idx >= NB * NHEAD) return;
  int n = idx >> 2, h = idx & 3;
  float ss = 0.0f, dd = 0.0f;
  for (int f = 0; f < 64; ++f) {
    float xv = xh[(size_t)n * NE + h * 64 + f];
    ss += xv * asrc[h * 64 + f];
    dd += xv * adst[h * 64 + f];
  }
  sb[idx] = ss;
  db[idx] = dd;
}

__global__ __launch_bounds__(256) void k_gat_init(
    float* __restrict__ emax, float* __restrict__ den,
    float* __restrict__ agg) {
  int idx = blockIdx.x * 256 + threadIdx.x;
  if (idx < NB * NHEAD) { emax[idx] = -1.0e30f; den[idx] = 0.0f; }
  if (idx < NB * NE)    agg[idx] = 0.0f;
}

__global__ __launch_bounds__(256) void k_gat_edge_max(
    const float* __restrict__ sb, const float* __restrict__ db,
    const int* __restrict__ eidx, float* __restrict__ emax) {
  int idx = blockIdx.x * 256 + threadIdx.x;
  if (idx >= NEDGE * NHEAD) return;
  int e = idx >> 2, h = idx & 3;
  int s = eidx[e], d = eidx[NEDGE + e];
  float v = leaky02(sb[s * NHEAD + h] + db[d * NHEAD + h]);
  atomicMaxF(&emax[d * NHEAD + h], v);
}

__global__ __launch_bounds__(256) void k_gat_edge_sum(
    const float* __restrict__ sb, const float* __restrict__ db,
    const int* __restrict__ eidx, const float* __restrict__ emax,
    float* __restrict__ den) {
  int idx = blockIdx.x * 256 + threadIdx.x;
  if (idx >= NEDGE * NHEAD) return;
  int e = idx >> 2, h = idx & 3;
  int s = eidx[e], d = eidx[NEDGE + e];
  float v = leaky02(sb[s * NHEAD + h] + db[d * NHEAD + h]);
  atomicAdd(&den[d * NHEAD + h], __expf(v - emax[d * NHEAD + h]));
}

__global__ __launch_bounds__(256) void k_gat_edge_agg(
    const float* __restrict__ sb, const float* __restrict__ db,
    const int* __restrict__ eidx, const float* __restrict__ emax,
    const float* __restrict__ den, const float* __restrict__ xh,
    float* __restrict__ agg) {
  int idx = blockIdx.x * 256 + threadIdx.x;
  if (idx >= NEDGE * NHEAD) return;
  int e = idx >> 2, h = idx & 3;
  int s = eidx[e], d = eidx[NEDGE + e];
  float v = leaky02(sb[s * NHEAD + h] + db[d * NHEAD + h]);
  float coef = __expf(v - emax[d * NHEAD + h]) /
               (den[d * NHEAD + h] + 1e-16f);
  const float* srow = xh + (size_t)s * NE + h * 64;
  float* drow = agg + (size_t)d * NE + h * 64;
  for (int f = 0; f < 64; ++f)
    atomicAdd(&drow[f], srow[f] * coef);
}

// ---------------------------------------------------------------------------
// launch
// ---------------------------------------------------------------------------
extern "C" void kernel_launch(void* const* d_in, const int* in_sizes, int n_in,
                              void* d_out, int out_size, void* d_ws,
                              size_t ws_size, hipStream_t stream) {
  const float* x       = (const float*)d_in[0];
  const int*   lengths = (const int*)d_in[1];
  const int*   eidx    = (const int*)d_in[2];
  const float* in_w    = (const float*)d_in[3];
  const float* in_b    = (const float*)d_in[4];
  const float* wqkv[2] = {(const float*)d_in[5], (const float*)d_in[9]};
  const float* bqkv[2] = {(const float*)d_in[6], (const float*)d_in[10]};
  const float* wo[2]   = {(const float*)d_in[7], (const float*)d_in[11]};
  const float* bo[2]   = {(const float*)d_in[8], (const float*)d_in[12]};
  const float* lng[2]  = {(const float*)d_in[13], (const float*)d_in[15]};
  const float* lnb[2]  = {(const float*)d_in[14], (const float*)d_in[16]};
  const float* gw[2]   = {(const float*)d_in[17], (const float*)d_in[21]};
  const float* gas[2]  = {(const float*)d_in[18], (const float*)d_in[22]};
  const float* gad[2]  = {(const float*)d_in[19], (const float*)d_in[23]};
  const float* gbi[2]  = {(const float*)d_in[20], (const float*)d_in[24]};
  const float* lgg[2]  = {(const float*)d_in[25], (const float*)d_in[27]};
  const float* lgb[2]  = {(const float*)d_in[26], (const float*)d_in[28]};
  const float* fcw     = (const float*)d_in[29];
  const float* fcb     = (const float*)d_in[30];

  const int ML = NB * NL;  // 131072 rows

  // ---- workspace carve (f16 activations + f16 weights + f32 GAT) --------
  _Float16* hp = (_Float16*)d_ws;
  _Float16* x16    = hp; hp += (size_t)ML * NIN;        // 8.4M
  _Float16* fh16   = hp; hp += (size_t)ML * NE;         // 33.5M
  _Float16* qkv16  = hp; hp += (size_t)ML * NQKV;       // 100.7M (also proj)
  _Float16* att16  = hp; hp += (size_t)ML * NE;         // 33.5M
  _Float16* inw16  = hp; hp += (size_t)NIN * NE;
  _Float16* wqkv16[2]; wqkv16[0] = hp; hp += (size_t)NE * NQKV;
                       wqkv16[1] = hp; hp += (size_t)NE * NQKV;
  _Float16* wo16[2];   wo16[0]   = hp; hp += (size_t)NE * NE;
                       wo16[1]   = hp; hp += (size_t)NE * NE;
  _Float16* gw16[2];   gw16[0]   = hp; hp += (size_t)NE * NE;
                       gw16[1]   = hp; hp += (size_t)NE * NE;
  _Float16* fcw16  = hp; hp += (size_t)NE * 16;
  _Float16* pooled = hp; hp += (size_t)NB * NE;
  _Float16* g116   = hp; hp += (size_t)NB * NE;
  _Float16* g216   = hp; hp += (size_t)NB * NE;
  float* fp = (float*)hp;
  float* xh   = fp; fp += (size_t)NB * NE;
  float* sb   = fp; fp += (size_t)NB * NHEAD;
  float* db   = fp; fp += (size_t)NB * NHEAD;
  float* emax = fp; fp += (size_t)NB * NHEAD;
  float* den  = fp; fp += (size_t)NB * NHEAD;
  float* agg  = fp; fp += (size_t)NB * NE;

  dim3 blk(256);
  auto cvt = [&](const float* s, _Float16* d, int n) {
    k_cvt<<<dim3((n + 255) / 256), blk, 0, stream>>>(s, d, n);
  };

  // ---- one-time (per launch) f32 -> f16 conversions ---------------------
  cvt(x, x16, ML * NIN);
  cvt(in_w, inw16, NIN * NE);
  for (int l = 0; l < 2; ++l) {
    cvt(wqkv[l], wqkv16[l], NE * NQKV);
    cvt(wo[l],   wo16[l],   NE * NE);
    cvt(gw[l],   gw16[l],   NE * NE);
  }
  cvt(fcw, fcw16, NE * 16);

  // ---- input projection -------------------------------------------------
  k_gemm16<false, true><<<dim3(ML / 256, NE / 64), blk, 0, stream>>>(
      x16, inw16, in_b, fh16, ML, NE, NIN);

  // ---- two transformer encoder layers -----------------------------------
  for (int l = 0; l < 2; ++l) {
    k_gemm16<false, true><<<dim3(ML / 256, NQKV / 64), blk, 0, stream>>>(
        fh16, wqkv16[l], bqkv[l], qkv16, ML, NQKV, NE);
    k_attention<<<dim3(NB * NHEAD), blk, 0, stream>>>(qkv16, lengths, att16);
    k_gemm16<false, true><<<dim3(ML / 256, NE / 64), blk, 0, stream>>>(
        att16, wo16[l], bo[l], qkv16, ML, NE, NE);  // proj -> qkv16 (reuse)
    k_res_ln<<<dim3(ML / 8), blk, 0, stream>>>(
        qkv16, nullptr, fh16, nullptr, lng[l], lnb[l], fh16, ML, 0);
  }

  // ---- masked mean pooling ----------------------------------------------
  k_pool<<<dim3(NB), blk, 0, stream>>>(fh16, lengths, pooled);

  // ---- two GAT layers ---------------------------------------------------
  const _Float16* gin = pooled;
  for (int l = 0; l < 2; ++l) {
    _Float16* gout = l ? g216 : g116;
    k_gemm16<false, false><<<dim3(NB / 256, NE / 64), blk, 0, stream>>>(
        gin, gw16[l], nullptr, xh, NB, NE, NE);
    k_gat_scores<<<dim3((NB * NHEAD + 255) / 256), blk, 0, stream>>>(
        xh, gas[l], gad[l], sb, db);
    k_gat_init<<<dim3((NB * NE + 255) / 256), blk, 0, stream>>>(emax, den, agg);
    k_gat_edge_max<<<dim3((NEDGE * NHEAD + 255) / 256), blk, 0, stream>>>(
        sb, db, eidx, emax);
    k_gat_edge_sum<<<dim3((NEDGE * NHEAD + 255) / 256), blk, 0, stream>>>(
        sb, db, eidx, emax, den);
    k_gat_edge_agg<<<dim3((NEDGE * NHEAD + 255) / 256), blk, 0, stream>>>(
        sb, db, eidx, emax, den, xh, agg);
    k_res_ln<<<dim3(NB / 8), blk, 0, stream>>>(
        nullptr, agg, nullptr, gbi[l], lgg[l], lgb[l], gout, NB, 1);
    gin = gout;
  }

  // ---- final FC ---------------------------------------------------------
  k_gemm16<true, false><<<dim3(NB / 256, 1), blk, 0, stream>>>(
      gin, fcw16, fcb, (float*)d_out, NB, 16, NE);
}
